// MultiHeadAttention_14001593385669
// MI455X (gfx1250) — compile-verified, matrix operand order checked by reading, not run on hardware
//
#include <hip/hip_runtime.h>
#include <hip/hip_bf16.h>

#define B_  8
#define S_  1024
#define D_  1024
#define H_  16
#define DK_ 64

typedef _Float16 half16 __attribute__((ext_vector_type(16)));
typedef _Float16 half8  __attribute__((ext_vector_type(8)));
typedef _Float16 half4  __attribute__((ext_vector_type(4)));
typedef float    float8 __attribute__((ext_vector_type(8)));
typedef __attribute__((address_space(3))) _Float16 lds_f16;

__device__ __forceinline__ float8 f8zero() {
  float8 z = {0.f,0.f,0.f,0.f,0.f,0.f,0.f,0.f};
  return z;
}

__device__ __forceinline__ float8 wmma_f16(half16 a, half16 b, float8 c) {
  // D = A(16x32) * B(32x16) + C, f32 accumulate
  return __builtin_amdgcn_wmma_f32_16x16x32_f16(false, a, false, b, (short)0, c, false, false);
}

// Async 16-byte global -> LDS copy (per-lane addresses), tracked by ASYNCcnt.
__device__ __forceinline__ void async_g2l_b128(unsigned lds_off, const _Float16* gsrc) {
  asm volatile("global_load_async_to_lds_b128 %0, %1, off"
               :: "v"(lds_off), "v"((unsigned long long)(uintptr_t)gsrc)
               : "memory");
}
__device__ __forceinline__ void wait_asynccnt0() {
  asm volatile("s_wait_asynccnt 0x0" ::: "memory");
}
__device__ __forceinline__ void wait_dscnt0() {
  asm volatile("s_wait_dscnt 0x0" ::: "memory");
}

// ---- WMMA fragment loaders (ISA 7.12.2 layouts, wave32) -------------------
// A fragment 16x32 f16 from row-major src (ldm halves/row).
__device__ __forceinline__ half16 load_a_frag(const _Float16* src, int ldm) {
  const int lane = threadIdx.x & 31;
  const int row  = lane & 15;
  const int koff = (lane >> 4) << 3;
  const _Float16* p0 = src + row * ldm + koff;
  const _Float16* p1 = p0 + 16;
  half16 a;
#pragma unroll
  for (int i = 0; i < 8; ++i) a[i] = p0[i];
#pragma unroll
  for (int i = 0; i < 8; ++i) a[8 + i] = p1[i];
  return a;
}

// B fragment 32x16 where element (k,n) lives at src[n*ldn + k] (k contiguous).
__device__ __forceinline__ half16 load_b_frag_kc(const _Float16* src, int ldn) {
  const int lane = threadIdx.x & 31;
  const int n    = lane & 15;
  const int koff = (lane >> 4) << 3;
  const _Float16* p0 = src + n * ldn + koff;
  const _Float16* p1 = p0 + 16;
  half16 b;
#pragma unroll
  for (int i = 0; i < 8; ++i) b[i] = p0[i];
#pragma unroll
  for (int i = 0; i < 8; ++i) b[8 + i] = p1[i];
  return b;
}

// ---- Kernel 1: per-head input projection  Out[b,h,s,:] = (X[b,s,:] @ W[h] + bias[h]) * scale
__global__ __launch_bounds__(128)
void proj_kernel(const float* __restrict__ X,    // [B,S,D]
                 const float* __restrict__ W,    // [H,D,DK]
                 const float* __restrict__ bias, // [H,DK]
                 _Float16* __restrict__ Out,     // [B*H,S,DK] f16
                 float scale) {
  __shared__ _Float16 As[64 * 32];   // [row][k]
  __shared__ _Float16 Bs[64 * 32];   // n-major: [n][k]
  const int tid  = threadIdx.x;
  const int wave = tid >> 5;
  const int lane = tid & 31;
  const int bh   = blockIdx.y;
  const int b = bh >> 4, h = bh & 15;
  const int rowbase = blockIdx.x * 64;

  const float* Xb = X + (size_t)b * S_ * D_ + (size_t)rowbase * D_;
  const float* Wh = W + (size_t)h * D_ * DK_;
  _Float16*    Ob = Out + (size_t)bh * S_ * DK_;

  float8 acc0 = f8zero(), acc1 = f8zero(), acc2 = f8zero(), acc3 = f8zero();

  for (int kk = 0; kk < D_; kk += 32) {
    __syncthreads();
    // A tile: 64x32 f32 -> f16 ; 512 float4 slots, 4 per thread, b128 loads.
#pragma unroll
    for (int i = 0; i < 4; ++i) {
      int slot = tid + i * 128;            // 0..511
      int r = slot >> 3;                   // row 0..63
      int c = (slot & 7) << 2;             // col 0,4,...,28
      float4 v = *(const float4*)(Xb + (size_t)r * D_ + kk + c);
      half4 hv = {(_Float16)v.x, (_Float16)v.y, (_Float16)v.z, (_Float16)v.w};
      *(half4*)(As + r * 32 + c) = hv;
    }
    // B tile: 32x64 f32, stored n-major (transposed) for k-contiguous frags.
#pragma unroll
    for (int i = 0; i < 4; ++i) {
      int slot = tid + i * 128;            // 0..511
      int r = slot >> 4;                   // k row 0..31
      int c = (slot & 15) << 2;            // n col 0,4,...,60
      float4 v = *(const float4*)(Wh + (size_t)(kk + r) * DK_ + c);
      Bs[(c + 0) * 32 + r] = (_Float16)v.x;
      Bs[(c + 1) * 32 + r] = (_Float16)v.y;
      Bs[(c + 2) * 32 + r] = (_Float16)v.z;
      Bs[(c + 3) * 32 + r] = (_Float16)v.w;
    }
    __syncthreads();

    half16 af = load_a_frag(As + wave * 16 * 32, 32);
    acc0 = wmma_f16(af, load_b_frag_kc(Bs +  0 * 32, 32), acc0);
    acc1 = wmma_f16(af, load_b_frag_kc(Bs + 16 * 32, 32), acc1);
    acc2 = wmma_f16(af, load_b_frag_kc(Bs + 32 * 32, 32), acc2);
    acc3 = wmma_f16(af, load_b_frag_kc(Bs + 48 * 32, 32), acc3);
  }

  const int n_l = lane & 15;
  const int rb  = (lane >> 4) << 3;
  const float* bh_bias = bias + h * DK_;
  float8 accs[4] = {acc0, acc1, acc2, acc3};
#pragma unroll
  for (int nt = 0; nt < 4; ++nt) {
    int n = nt * 16 + n_l;
    float bv = bh_bias[n];
#pragma unroll
    for (int r = 0; r < 8; ++r) {
      int m = rowbase + wave * 16 + rb + r;
      Ob[(size_t)m * DK_ + n] = (_Float16)((accs[nt][r] + bv) * scale);
    }
  }
}

// ---- Kernel 2: flash-style masked attention; block = 4 waves = 64 q rows of one (b,h)
__global__ __launch_bounds__(128)
void attn_kernel(const _Float16* __restrict__ Q,   // [B*H,S,DK] (pre-scaled by 1/sqrt(DK))
                 const _Float16* __restrict__ Km,  // [B*H,S,DK]
                 const _Float16* __restrict__ Vm,  // [B*H,S,DK]
                 const int* __restrict__ mask0,    // [B,S,S]
                 const int* __restrict__ mask1,    // [B,S,S]
                 _Float16* __restrict__ AOut) {    // [B,S,H*DK] f16
  __shared__ _Float16 Kt[32 * 64];     // [s_local][dk]     (scores B-frags, k-contig)
  __shared__ _Float16 Vt[64 * 32];     // [dk][s_local]     (PV B-frags, k-contig)
  __shared__ _Float16 Ps[4 * 16 * 32]; // per-wave P transpose staging
  const int tid  = threadIdx.x;
  const int wave = tid >> 5;
  const int lane = tid & 31;
  const int n_l  = lane & 15;
  const int rb   = (lane >> 4) << 3;
  const int bh   = blockIdx.y;
  const int b = bh >> 4, h = bh & 15;
  const int q0 = blockIdx.x * 64 + wave * 16;

  const _Float16* Qbh = Q  + (size_t)bh * S_ * DK_;
  const _Float16* Kbh = Km + (size_t)bh * S_ * DK_;
  const _Float16* Vbh = Vm + (size_t)bh * S_ * DK_;
  const int* m0 = mask0 + (size_t)b * S_ * S_;
  const int* m1 = mask1 + (size_t)b * S_ * S_;
  _Float16* Pw = Ps + wave * (16 * 32);
  const unsigned kt_base = (unsigned)(uintptr_t)(lds_f16*)Kt;  // LDS byte offset of Kt

  half16 qf0 = load_a_frag(Qbh + (size_t)q0 * DK_, DK_);        // k 0..31
  half16 qf1 = load_a_frag(Qbh + (size_t)q0 * DK_ + 32, DK_);   // k 32..63

  const float NEG_INF = -__builtin_inff();
  float mrun[8], lrun[8];
  float8 o0 = f8zero(), o1 = f8zero(), o2 = f8zero(), o3 = f8zero();
#pragma unroll
  for (int r = 0; r < 8; ++r) { mrun[r] = NEG_INF; lrun[r] = 0.f; }

  for (int s0 = 0; s0 < S_; s0 += 32) {
    __syncthreads();   // all waves done reading previous Kt/Vt

    // K tile: 32x64 f16 = 4 KB, async global->LDS, 2x16B per thread.
    {
      const _Float16* gk = Kbh + (size_t)s0 * DK_;
      async_g2l_b128(kt_base + (unsigned)tid * 16u,          gk + tid * 8);
      async_g2l_b128(kt_base + (unsigned)(tid + 128) * 16u,  gk + (tid + 128) * 8);
    }
    // V tile transposed at staging: Vt[dk][s_local].
    {
      const _Float16* gv = Vbh + (size_t)s0 * DK_;
#pragma unroll
      for (int i = 0; i < 2; ++i) {
        int slot = tid + i * 128;          // 256 slots of 8 halves
        int srow = slot >> 3;              // s_local 0..31
        int dk0  = (slot & 7) << 3;        // dk 0,8,...,56
        half8 d = *(const half8*)(gv + slot * 8);
#pragma unroll
        for (int j = 0; j < 8; ++j) Vt[(dk0 + j) * 32 + srow] = d[j];
      }
    }
    wait_asynccnt0();   // our async K copies landed in LDS
    __syncthreads();    // staged tiles visible block-wide

    // scores: 16x32 tile of q-rows x keys (two 16x16 WMMA accumulators)
    float8 sc0 = f8zero(), sc1 = f8zero();
    sc0 = wmma_f16(qf0, load_b_frag_kc(Kt,                64), sc0);
    sc0 = wmma_f16(qf1, load_b_frag_kc(Kt + 32,           64), sc0);
    sc1 = wmma_f16(qf0, load_b_frag_kc(Kt + 16 * 64,      64), sc1);
    sc1 = wmma_f16(qf1, load_b_frag_kc(Kt + 16 * 64 + 32, 64), sc1);

    // pre-softmax key-padding mask
#pragma unroll
    for (int r = 0; r < 8; ++r) {
      int m = q0 + rb + r;
      sc0[r] = (m0[(size_t)m * S_ + s0 + n_l]      == 0) ? NEG_INF : sc0[r];
      sc1[r] = (m0[(size_t)m * S_ + s0 + 16 + n_l] == 0) ? NEG_INF : sc1[r];
    }

    // row max over this 32-key chunk (lanes 0-15: rows 0-7, lanes 16-31: rows 8-15)
    float rmax[8];
#pragma unroll
    for (int r = 0; r < 8; ++r) rmax[r] = fmaxf(sc0[r], sc1[r]);
#pragma unroll
    for (int off = 8; off >= 1; off >>= 1)
#pragma unroll
      for (int r = 0; r < 8; ++r)
        rmax[r] = fmaxf(rmax[r], __shfl_xor(rmax[r], off, 32));

    float mcl[8], alpha[8], rsum[8];
#pragma unroll
    for (int r = 0; r < 8; ++r) {
      float mnew = fmaxf(mrun[r], rmax[r]);
      mcl[r]   = (mnew == NEG_INF) ? 0.f : mnew;
      alpha[r] = (mrun[r] == NEG_INF) ? 0.f : __expf(mrun[r] - mcl[r]);
      mrun[r]  = mnew;
      rsum[r]  = 0.f;
    }

    // p = exp(s - m); denominator uses pre-mask1 p; PV uses mask1*p
#pragma unroll
    for (int r = 0; r < 8; ++r) {
      int m = q0 + rb + r;
      float p0 = __expf(sc0[r] - mcl[r]);
      float p1 = __expf(sc1[r] - mcl[r]);
      rsum[r] += p0 + p1;
      float w0 = m1[(size_t)m * S_ + s0 + n_l]      ? p0 : 0.f;
      float w1 = m1[(size_t)m * S_ + s0 + 16 + n_l] ? p1 : 0.f;
      Pw[(rb + r) * 32 + n_l]      = (_Float16)w0;
      Pw[(rb + r) * 32 + 16 + n_l] = (_Float16)w1;
    }

#pragma unroll
    for (int off = 8; off >= 1; off >>= 1)
#pragma unroll
      for (int r = 0; r < 8; ++r)
        rsum[r] += __shfl_xor(rsum[r], off, 32);

#pragma unroll
    for (int r = 0; r < 8; ++r) {
      lrun[r] = lrun[r] * alpha[r] + rsum[r];
      o0[r] *= alpha[r]; o1[r] *= alpha[r]; o2[r] *= alpha[r]; o3[r] *= alpha[r];
    }

    wait_dscnt0();   // P tile visible wave-wide before transposed re-read

    half16 pf = load_a_frag(Pw, 32);
    o0 = wmma_f16(pf, load_b_frag_kc(Vt +  0 * 32, 32), o0);
    o1 = wmma_f16(pf, load_b_frag_kc(Vt + 16 * 32, 32), o1);
    o2 = wmma_f16(pf, load_b_frag_kc(Vt + 32 * 32, 32), o2);
    o3 = wmma_f16(pf, load_b_frag_kc(Vt + 48 * 32, 32), o3);
  }

  float inv[8];
#pragma unroll
  for (int r = 0; r < 8; ++r) inv[r] = (lrun[r] > 0.f) ? 1.f / lrun[r] : 0.f;

  float8 oo[4] = {o0, o1, o2, o3};
#pragma unroll
  for (int nt = 0; nt < 4; ++nt)
#pragma unroll
    for (int r = 0; r < 8; ++r) {
      int m = q0 + rb + r;
      AOut[((size_t)(b * S_ + m)) * D_ + h * DK_ + nt * 16 + n_l] =
          (_Float16)(oo[nt][r] * inv[r]);
    }
}

// ---- Kernel 3: output projection  Out = A(f16) @ Wo + bo  (f32 out)
__global__ __launch_bounds__(128)
void outproj_kernel(const _Float16* __restrict__ A,  // [B*S, D] f16
                    const float* __restrict__ W,     // [D, D]
                    const float* __restrict__ bias,  // [D]
                    float* __restrict__ Out) {       // [B*S, D]
  __shared__ _Float16 As[64 * 32];
  __shared__ _Float16 Bs[64 * 32];  // n-major
  const int tid  = threadIdx.x;
  const int wave = tid >> 5;
  const int lane = tid & 31;
  const int rowbase = blockIdx.y * 64;
  const int nbase   = blockIdx.x * 64;
  const _Float16* Ab = A + (size_t)rowbase * D_;

  float8 acc0 = f8zero(), acc1 = f8zero(), acc2 = f8zero(), acc3 = f8zero();

  for (int kk = 0; kk < D_; kk += 32) {
    __syncthreads();
    // A tile: f16 copy, 256 x 16B slots, 2 per thread.
#pragma unroll
    for (int i = 0; i < 2; ++i) {
      int slot = tid + i * 128;           // 0..255
      int r = slot >> 2;                  // row 0..63
      int c = (slot & 3) << 3;            // col 0,8,16,24
      *(half8*)(As + r * 32 + c) = *(const half8*)(Ab + (size_t)r * D_ + kk + c);
    }
    // B tile: 32x64 f32 -> n-major f16.
#pragma unroll
    for (int i = 0; i < 4; ++i) {
      int slot = tid + i * 128;           // 0..511
      int r = slot >> 4;                  // k row 0..31
      int c = (slot & 15) << 2;           // n col 0,4,...,60
      float4 v = *(const float4*)(W + (size_t)(kk + r) * D_ + nbase + c);
      Bs[(c + 0) * 32 + r] = (_Float16)v.x;
      Bs[(c + 1) * 32 + r] = (_Float16)v.y;
      Bs[(c + 2) * 32 + r] = (_Float16)v.z;
      Bs[(c + 3) * 32 + r] = (_Float16)v.w;
    }
    __syncthreads();

    half16 af = load_a_frag(As + wave * 16 * 32, 32);
    acc0 = wmma_f16(af, load_b_frag_kc(Bs +  0 * 32, 32), acc0);
    acc1 = wmma_f16(af, load_b_frag_kc(Bs + 16 * 32, 32), acc1);
    acc2 = wmma_f16(af, load_b_frag_kc(Bs + 32 * 32, 32), acc2);
    acc3 = wmma_f16(af, load_b_frag_kc(Bs + 48 * 32, 32), acc3);
  }

  const int n_l = lane & 15;
  const int rb  = (lane >> 4) << 3;
  float8 accs[4] = {acc0, acc1, acc2, acc3};
#pragma unroll
  for (int nt = 0; nt < 4; ++nt) {
    int n = nbase + nt * 16 + n_l;
    float bv = bias[n];
#pragma unroll
    for (int r = 0; r < 8; ++r) {
      int m = rowbase + wave * 16 + rb + r;
      Out[(size_t)m * D_ + n] = accs[nt][r] + bv;
    }
  }
}

extern "C" void kernel_launch(void* const* d_in, const int* in_sizes, int n_in,
                              void* d_out, int out_size, void* d_ws, size_t ws_size,
                              hipStream_t stream) {
  const float* query = (const float*)d_in[0];
  const float* key   = (const float*)d_in[1];
  const float* value = (const float*)d_in[2];
  const int*   mask0 = (const int*)d_in[3];
  const int*   mask1 = (const int*)d_in[4];
  const float* Wq    = (const float*)d_in[5];
  const float* bq    = (const float*)d_in[6];
  const float* Wk    = (const float*)d_in[7];
  const float* bk    = (const float*)d_in[8];
  const float* Wv    = (const float*)d_in[9];
  const float* bv    = (const float*)d_in[10];
  const float* Wo    = (const float*)d_in[11];
  const float* bo    = (const float*)d_in[12];
  float* out = (float*)d_out;

  const size_t perTensor = (size_t)B_ * H_ * S_ * DK_;   // 8 Mi halves = 16 MB
  _Float16* Qh = (_Float16*)d_ws;
  _Float16* Kh = Qh + perTensor;
  _Float16* Vh = Kh + perTensor;
  _Float16* Ao = Vh + perTensor;

  dim3 blk(128);
  dim3 gproj(S_ / 64, B_ * H_);
  proj_kernel<<<gproj, blk, 0, stream>>>(query, Wq, bq, Qh, 0.125f); // fold 1/sqrt(64)
  proj_kernel<<<gproj, blk, 0, stream>>>(key,   Wk, bk, Kh, 1.0f);
  proj_kernel<<<gproj, blk, 0, stream>>>(value, Wv, bv, Vh, 1.0f);
  attn_kernel<<<dim3(S_ / 64, B_ * H_), blk, 0, stream>>>(Qh, Kh, Vh, mask0, mask1, Ao);
  outproj_kernel<<<dim3(D_ / 64, (B_ * S_) / 64), blk, 0, stream>>>(Ao, Wo, bo, out);

  (void)in_sizes; (void)n_in; (void)out_size; (void)ws_size;
}
